// GraphTransformerGather_60713657696828
// MI455X (gfx1250) — matrix operand order repaired and checked by reading, not compiled, not run on hardware
//
#include <hip/hip_runtime.h>
#include <hip/hip_bf16.h>

// ---------------- problem constants ----------------
constexpr int N_NODES = 6144;
constexpr int N_EDGES = 196608;
constexpr int D_MODEL = 128;
constexpr int EDGE_DIM = 16;
constexpr int N_HEADS = 4;
constexpr int DH = 32;          // head dim == WMMA K for bf16 (16x16x32)
constexpr int FF_DIM = 512;
constexpr int N_LAYERS = 3;

typedef __attribute__((ext_vector_type(16))) __bf16 v16bf;
typedef __attribute__((ext_vector_type(8)))  float  v8f;

union F16x16 { uint4 u[2]; v16bf v; };

// ---------------- CDNA5 async global->LDS copy (ASYNCcnt-tracked DMA) ----------------
__device__ __forceinline__ void async_ld_b128(unsigned lds_off, const void* gaddr) {
  asm volatile("global_load_async_to_lds_b128 %0, %1, off"
               :: "v"(lds_off), "v"(gaddr) : "memory");
}
__device__ __forceinline__ void wait_async0() {
  asm volatile("s_wait_asynccnt 0x0" ::: "memory");
}
__device__ __forceinline__ unsigned lds_off_of(const void* p) {
  return (unsigned)(uintptr_t)p;   // flat LDS addr: low 32 bits are the LDS offset
}

// ---------------- WMMA fragment loaders (2 x ds_load_b128 each) ----------------
// A-matrix 16x32 bf16, row-major LDS tile, ld elements/row (ld*2 % 16 == 0).
__device__ __forceinline__ v16bf lda_frag(const __bf16* tile, int ld, int lane) {
  const int half = lane >> 4, m = lane & 15;
  const __bf16* p = tile + m * ld + half * 8;
  F16x16 t;
  t.u[0] = *(const uint4*)(p);        // k = half*8 + 0..7
  t.u[1] = *(const uint4*)(p + 16);   // k = 16 + half*8 + 0..7
  return t.v;
}
// B-matrix 32x16 bf16 from a column-major LDS tile [n][k], ld elements per n-row.
__device__ __forceinline__ v16bf ldb_frag(const __bf16* tile, int ld, int lane) {
  const int half = lane >> 4, n = lane & 15;
  const __bf16* p = tile + n * ld + half * 8;
  F16x16 t;
  t.u[0] = *(const uint4*)(p);
  t.u[1] = *(const uint4*)(p + 16);
  return t.v;
}

__device__ __forceinline__ v8f wmma_bf16(v16bf a, v16bf b, v8f c) {
  return __builtin_amdgcn_wmma_f32_16x16x32_bf16(false, a, false, b, (short)0, c, false, false);
}

// ---------------- tiny utility kernels ----------------
__global__ void zero_f32_kernel(float* p, int n) {
  int i = blockIdx.x * 256 + threadIdx.x;
  if (i < n) p[i] = 0.0f;
}

// convert f32 [K][Nw] -> bf16 transposed [Nw][K]
__global__ void convT_bf16_kernel(const float* __restrict__ s, __bf16* __restrict__ d,
                                  int K, int Nw) {
  int i = blockIdx.x * 256 + threadIdx.x;
  if (i < K * Nw) {
    int k = i / Nw, n = i - k * Nw;
    d[(size_t)n * K + k] = (__bf16)s[i];
  }
}

// edge_bias = edge_feats @ We + be; scatter-add by dst -> attn_bias[h][n]
__global__ void edge_kernel(const float* __restrict__ ef, const float* __restrict__ We,
                            const float* __restrict__ be, const int* __restrict__ dst,
                            float* __restrict__ ab) {
  int e = blockIdx.x * 256 + threadIdx.x;
  if (e >= N_EDGES) return;
  const float4* f4 = (const float4*)(ef + (size_t)e * EDGE_DIM);
  float acc[N_HEADS];
#pragma unroll
  for (int h = 0; h < N_HEADS; ++h) acc[h] = be[h];
#pragma unroll
  for (int q = 0; q < 4; ++q) {
    float4 fv = f4[q];
    const float* fp = (const float*)&fv;
#pragma unroll
    for (int j = 0; j < 4; ++j)
#pragma unroll
      for (int h = 0; h < N_HEADS; ++h)
        acc[h] += fp[j] * We[(q * 4 + j) * N_HEADS + h];
  }
  int d = dst[e];
#pragma unroll
  for (int h = 0; h < N_HEADS; ++h) atomicAdd(&ab[h * N_NODES + d], acc[h]);
}

// LayerNorm over last dim (128). One wave per row, 4 rows per 128-thread block.
__global__ void __launch_bounds__(128) ln_kernel(const float* __restrict__ xa,
                                                 const float* __restrict__ xb,
                                                 const float* __restrict__ sc,
                                                 const float* __restrict__ bi,
                                                 float* __restrict__ outf,
                                                 __bf16* __restrict__ outb) {
  const int row  = blockIdx.x * 4 + (threadIdx.x >> 5);
  const int lane = threadIdx.x & 31;
  const float4* ra4 = (const float4*)(xa + (size_t)row * D_MODEL);
  float4 va = ra4[lane];
  float v[4] = {va.x, va.y, va.z, va.w};
  if (xb) {
    float4 vb = ((const float4*)(xb + (size_t)row * D_MODEL))[lane];
    v[0] += vb.x; v[1] += vb.y; v[2] += vb.z; v[3] += vb.w;
  }
  float sum = v[0] + v[1] + v[2] + v[3];
#pragma unroll
  for (int m = 1; m < 32; m <<= 1) sum += __shfl_xor(sum, m, 32);
  const float mean = sum * (1.0f / D_MODEL);
  float vs = 0.0f;
#pragma unroll
  for (int i = 0; i < 4; ++i) { float d = v[i] - mean; vs += d * d; }
#pragma unroll
  for (int m = 1; m < 32; m <<= 1) vs += __shfl_xor(vs, m, 32);
  const float rinv = rsqrtf(vs * (1.0f / D_MODEL) + 1e-5f);
  float4 s4 = ((const float4*)sc)[lane];
  float4 b4 = ((const float4*)bi)[lane];
  const float* sp = (const float*)&s4;
  const float* bp = (const float*)&b4;
  float o[4];
#pragma unroll
  for (int i = 0; i < 4; ++i) o[i] = (v[i] - mean) * rinv * sp[i] + bp[i];
  if (outf) {
    float4 w = {o[0], o[1], o[2], o[3]};
    ((float4*)(outf + (size_t)row * D_MODEL))[lane] = w;
  }
  if (outb) {
    union { __bf16 h[4]; uint2 u; } w;
#pragma unroll
    for (int i = 0; i < 4; ++i) w.h[i] = (__bf16)o[i];
    ((uint2*)(outb + (size_t)row * D_MODEL))[lane] = w.u;
  }
}

// ---------------- WMMA GEMM: tile 128x64, K step 32, 8 waves, double-buffered DMA ----
// A: [M][K] bf16 row-major; Wt: [Nw][K] bf16 (pre-transposed weights).
// MODE 0: outb = (acc+bias)*alpha ; MODE 1: outb = relu(acc+bias) ; MODE 2: outf = acc+bias+res
__device__ __forceinline__ void gemm_stage(const __bf16* A, const __bf16* Wt,
                                           unsigned abase, unsigned bbase,
                                           int m0, int n0, int K, int kc, int tid) {
#pragma unroll
  for (int ii = 0; ii < 2; ++ii) {
    int i = tid + ii * 256;
    int r = i >> 2, c = (i & 3) * 8;
    async_ld_b128(abase + (unsigned)(r * 40 + c) * 2,
                  &A[(size_t)(m0 + r) * K + kc + c]);
  }
  int n = tid >> 2, ck = (tid & 3) * 8;
  async_ld_b128(bbase + (unsigned)(n * 40 + ck) * 2,
                &Wt[(size_t)(n0 + n) * K + kc + ck]);
}

template <int MODE>
__global__ void __launch_bounds__(256) gemm_kernel(const __bf16* __restrict__ A,
                                                   const __bf16* __restrict__ Wt,
                                                   const float* __restrict__ bias,
                                                   const float* __restrict__ res,
                                                   float* __restrict__ outf,
                                                   __bf16* __restrict__ outb,
                                                   int K, int Nw, float alpha) {
  const int m0 = blockIdx.x * 128;
  const int n0 = blockIdx.y * 64;
  const int tid = threadIdx.x, wave = tid >> 5, lane = tid & 31;
  const int half = lane >> 4, ln = lane & 15;

  __shared__ __attribute__((aligned(16))) __bf16 As[2][128 * 40];
  __shared__ __attribute__((aligned(16))) __bf16 Bs[2][64 * 40];
  const unsigned abase[2] = {lds_off_of(As[0]), lds_off_of(As[1])};
  const unsigned bbase[2] = {lds_off_of(Bs[0]), lds_off_of(Bs[1])};

  v8f acc[4] = {{}, {}, {}, {}};

  gemm_stage(A, Wt, abase[0], bbase[0], m0, n0, K, 0, tid);

  int idx = 0;
  for (int kc = 0; kc < K; kc += 32, idx ^= 1) {
    wait_async0();            // tile[idx] landed in LDS
    __syncthreads();          // visible to all waves; prior reads of tile[idx^1] done
    if (kc + 32 < K)          // overlap next DMA with this tile's WMMAs
      gemm_stage(A, Wt, abase[idx ^ 1], bbase[idx ^ 1], m0, n0, K, kc + 32, tid);

    v16bf a = lda_frag(As[idx] + wave * 16 * 40, 40, lane);
#pragma unroll
    for (int t = 0; t < 4; ++t) {
      v16bf b = ldb_frag(Bs[idx] + t * 16 * 40, 40, lane);
      acc[t] = wmma_bf16(a, b, acc[t]);
    }
  }

#pragma unroll
  for (int t = 0; t < 4; ++t) {
#pragma unroll
    for (int e = 0; e < 8; ++e) {
      int row = m0 + wave * 16 + e + 8 * half;
      int col = n0 + t * 16 + ln;
      float v = acc[t][e] + bias[col];
      if constexpr (MODE == 0) {
        outb[(size_t)row * Nw + col] = (__bf16)(v * alpha);
      } else if constexpr (MODE == 1) {
        outb[(size_t)row * Nw + col] = (__bf16)fmaxf(v, 0.0f);
      } else {
        outf[(size_t)row * Nw + col] = v + res[(size_t)row * Nw + col];
      }
    }
  }
}

// ---------------- flash attention: per (qtile of 64 rows, head) ----------------
// Q pre-scaled by 1/sqrt(DH). scores += bias[h][key]; online softmax; O = P@V.
// K tile via async DMA, V tile transposed manually; both double-buffered.
__global__ void __launch_bounds__(128) attn_kernel(const __bf16* __restrict__ Qg,
                                                   const __bf16* __restrict__ Kg,
                                                   const __bf16* __restrict__ Vg,
                                                   const float* __restrict__ bias,
                                                   __bf16* __restrict__ outg) {
  const int h  = blockIdx.y;
  const int q0 = blockIdx.x * 64;
  const int tid = threadIdx.x, wave = tid >> 5, lane = tid & 31;
  const int half = lane >> 4, ln = lane & 15;

  __shared__ __attribute__((aligned(16))) __bf16 Qs[64 * 40];        // [query][dh]
  __shared__ __attribute__((aligned(16))) __bf16 Ks[2][64 * 40];     // [key][dh]
  __shared__ __attribute__((aligned(16))) __bf16 Vs[2][32 * 72];     // [dh][key]
  __shared__ __attribute__((aligned(16))) __bf16 Ps[4][16 * 72];     // per-wave P
  const unsigned qbase = lds_off_of(Qs);
  const unsigned kbase[2] = {lds_off_of(Ks[0]), lds_off_of(Ks[1])};

  auto stage_k = [&](int b, int k0) {
#pragma unroll
    for (int ii = 0; ii < 2; ++ii) {
      int i = tid + ii * 128;
      int r = i >> 2, c = (i & 3) * 8;
      async_ld_b128(kbase[b] + (unsigned)(r * 40 + c) * 2,
                    &Kg[(size_t)(k0 + r) * D_MODEL + h * DH + c]);
    }
  };
  auto stage_v = [&](int b, int k0) {
#pragma unroll
    for (int ii = 0; ii < 8; ++ii) {
      int i = tid + ii * 128;
      int r = i >> 4, c = (i & 15) * 2;   // key r, dh c
      unsigned int vv =
          *(const unsigned int*)&Vg[(size_t)(k0 + r) * D_MODEL + h * DH + c];
      const __bf16* vp = (const __bf16*)&vv;
      Vs[b][(c + 0) * 72 + r] = vp[0];
      Vs[b][(c + 1) * 72 + r] = vp[1];
    }
  };

  // stage Q + first K/V tile
#pragma unroll
  for (int ii = 0; ii < 2; ++ii) {
    int i = tid + ii * 128;
    int r = i >> 2, c = (i & 3) * 8;
    async_ld_b128(qbase + (unsigned)(r * 40 + c) * 2,
                  &Qg[(size_t)(q0 + r) * D_MODEL + h * DH + c]);
  }
  stage_k(0, 0);
  stage_v(0, 0);
  wait_async0();
  __syncthreads();
  const v16bf aq = lda_frag(Qs + wave * 16 * 40, 40, lane);

  float m_i[8], l_i[8];
#pragma unroll
  for (int e = 0; e < 8; ++e) { m_i[e] = -3.0e38f; l_i[e] = 0.0f; }
  v8f o0 = {}, o1 = {};

  const float* brow = bias + (size_t)h * N_NODES;

  int idx = 0;
  for (int k0 = 0; k0 < N_NODES; k0 += 64, idx ^= 1) {
    if (k0 + 64 < N_NODES) {      // prefetch next tile while computing this one
      stage_k(idx ^ 1, k0 + 64);
      stage_v(idx ^ 1, k0 + 64);
    }

    // S = Q Kt
    v8f s[4];
#pragma unroll
    for (int t = 0; t < 4; ++t) {
      v16bf b = ldb_frag(Ks[idx] + t * 16 * 40, 40, lane);
      v8f z = {};
      s[t] = wmma_bf16(aq, b, z);
    }
#pragma unroll
    for (int t = 0; t < 4; ++t)
#pragma unroll
      for (int e = 0; e < 8; ++e) s[t][e] += brow[k0 + t * 16 + ln];

    // online softmax
    float rm[8];
#pragma unroll
    for (int e = 0; e < 8; ++e)
      rm[e] = fmaxf(fmaxf(s[0][e], s[1][e]), fmaxf(s[2][e], s[3][e]));
#pragma unroll
    for (int m = 1; m < 16; m <<= 1)
#pragma unroll
      for (int e = 0; e < 8; ++e) rm[e] = fmaxf(rm[e], __shfl_xor(rm[e], m, 32));

    float corr[8], rs[8];
#pragma unroll
    for (int e = 0; e < 8; ++e) {
      float mn = fmaxf(m_i[e], rm[e]);
      corr[e] = __expf(m_i[e] - mn);
      m_i[e] = mn;
      rs[e] = 0.0f;
    }
#pragma unroll
    for (int t = 0; t < 4; ++t)
#pragma unroll
      for (int e = 0; e < 8; ++e) {
        float pv = __expf(s[t][e] - m_i[e]);
        s[t][e] = pv;
        rs[e] += pv;
      }
#pragma unroll
    for (int m = 1; m < 16; m <<= 1)
#pragma unroll
      for (int e = 0; e < 8; ++e) rs[e] += __shfl_xor(rs[e], m, 32);
#pragma unroll
    for (int e = 0; e < 8; ++e) {
      l_i[e] = l_i[e] * corr[e] + rs[e];
      o0[e] *= corr[e];
      o1[e] *= corr[e];
    }

    // re-layout P (C-frag) -> bf16 A-frag via per-wave LDS
    __bf16* Pw = &Ps[wave][0];
#pragma unroll
    for (int t = 0; t < 4; ++t)
#pragma unroll
      for (int e = 0; e < 8; ++e)
        Pw[(e + 8 * half) * 72 + t * 16 + ln] = (__bf16)s[t][e];
    __syncthreads();

    v16bf a0 = lda_frag(Pw, 72, lane);        // keys 0..31
    v16bf a1 = lda_frag(Pw + 32, 72, lane);   // keys 32..63
    const __bf16* V = Vs[idx];
    v16bf b00 = ldb_frag(V, 72, lane);                 // k 0-31,  dh 0-15
    v16bf b01 = ldb_frag(V + 16 * 72, 72, lane);       // k 0-31,  dh 16-31
    v16bf b10 = ldb_frag(V + 32, 72, lane);            // k 32-63, dh 0-15
    v16bf b11 = ldb_frag(V + 16 * 72 + 32, 72, lane);  // k 32-63, dh 16-31
    o0 = wmma_bf16(a0, b00, o0);
    o0 = wmma_bf16(a1, b10, o0);
    o1 = wmma_bf16(a0, b01, o1);
    o1 = wmma_bf16(a1, b11, o1);

    // next tile ready + everyone done with this one
    wait_async0();
    __syncthreads();
  }

#pragma unroll
  for (int e = 0; e < 8; ++e) {
    int row = q0 + wave * 16 + e + 8 * half;
    float inv = 1.0f / l_i[e];
    outg[(size_t)row * D_MODEL + h * DH + ln]      = (__bf16)(o0[e] * inv);
    outg[(size_t)row * D_MODEL + h * DH + 16 + ln] = (__bf16)(o1[e] * inv);
  }
}

// ---------------- host orchestration ----------------
extern "C" void kernel_launch(void* const* d_in, const int* in_sizes, int n_in,
                              void* d_out, int out_size, void* d_ws, size_t ws_size,
                              hipStream_t stream) {
  (void)in_sizes; (void)n_in; (void)out_size; (void)ws_size;

  const float* node = (const float*)d_in[0];
  const float* ef   = (const float*)d_in[1];
  const float* We   = (const float*)d_in[2];
  const float* be   = (const float*)d_in[3];
  const int*   dst  = (const int*)d_in[4 + N_LAYERS * 18 + 1];  // src then dst

  char* ws = (char*)d_ws;
  size_t off = 0;
  auto carve = [&](size_t bytes) -> void* {
    void* p = ws + off;
    off = (off + bytes + 255) & ~(size_t)255;
    return p;
  };
  float*  ab   = (float*)carve((size_t)N_HEADS * N_NODES * 4);
  __bf16* xln  = (__bf16*)carve((size_t)N_NODES * D_MODEL * 2);
  __bf16* Qb   = (__bf16*)carve((size_t)N_NODES * D_MODEL * 2);
  __bf16* Kb   = (__bf16*)carve((size_t)N_NODES * D_MODEL * 2);
  __bf16* Vb   = (__bf16*)carve((size_t)N_NODES * D_MODEL * 2);
  __bf16* ao   = (__bf16*)carve((size_t)N_NODES * D_MODEL * 2);
  float*  t1   = (float*)carve((size_t)N_NODES * D_MODEL * 4);
  float*  t2   = (float*)carve((size_t)N_NODES * D_MODEL * 4);
  __bf16* x2b  = (__bf16*)carve((size_t)N_NODES * D_MODEL * 2);
  __bf16* ub   = (__bf16*)carve((size_t)N_NODES * FF_DIM * 2);
  float*  hA   = (float*)carve((size_t)N_NODES * D_MODEL * 4);
  float*  hB   = (float*)carve((size_t)N_NODES * D_MODEL * 4);
  __bf16* wqb  = (__bf16*)carve((size_t)D_MODEL * D_MODEL * 2);  // transposed [out][in]
  __bf16* wkb  = (__bf16*)carve((size_t)D_MODEL * D_MODEL * 2);
  __bf16* wvb  = (__bf16*)carve((size_t)D_MODEL * D_MODEL * 2);
  __bf16* wob  = (__bf16*)carve((size_t)D_MODEL * D_MODEL * 2);
  __bf16* w1b  = (__bf16*)carve((size_t)D_MODEL * FF_DIM * 2);
  __bf16* w2b  = (__bf16*)carve((size_t)FF_DIM * D_MODEL * 2);

  // edge bias -> attn_bias[H][N]
  zero_f32_kernel<<<(N_HEADS * N_NODES + 255) / 256, 256, 0, stream>>>(ab, N_HEADS * N_NODES);
  edge_kernel<<<(N_EDGES + 255) / 256, 256, 0, stream>>>(ef, We, be, dst, ab);

  const float qscale = 0.17677669529663689f;  // 1/sqrt(32)
  const float* hin = node;
  float* houts[N_LAYERS] = {hA, hB, (float*)d_out};

  const dim3 g_rows(N_NODES / 4);
  const dim3 g_gemm128(N_NODES / 128, D_MODEL / 64);
  const dim3 g_gemmff(N_NODES / 128, FF_DIM / 64);
  const dim3 g_attn(N_NODES / 64, N_HEADS);

  for (int l = 0; l < N_LAYERS; ++l) {
    const float* Wq = (const float*)d_in[4 + l * 18 + 0];
    const float* bq = (const float*)d_in[4 + l * 18 + 1];
    const float* Wk = (const float*)d_in[4 + l * 18 + 2];
    const float* bk = (const float*)d_in[4 + l * 18 + 3];
    const float* Wv = (const float*)d_in[4 + l * 18 + 4];
    const float* bv = (const float*)d_in[4 + l * 18 + 5];
    const float* Wo = (const float*)d_in[4 + l * 18 + 6];
    const float* bo = (const float*)d_in[4 + l * 18 + 7];
    const float* l1s = (const float*)d_in[4 + l * 18 + 8];
    const float* l1b = (const float*)d_in[4 + l * 18 + 9];
    const float* l2s = (const float*)d_in[4 + l * 18 + 10];
    const float* l2b = (const float*)d_in[4 + l * 18 + 11];
    const float* W1 = (const float*)d_in[4 + l * 18 + 12];
    const float* b1 = (const float*)d_in[4 + l * 18 + 13];
    const float* W2 = (const float*)d_in[4 + l * 18 + 14];
    const float* b2 = (const float*)d_in[4 + l * 18 + 15];
    const float* nsc = (const float*)d_in[4 + l * 18 + 16];
    const float* nbi = (const float*)d_in[4 + l * 18 + 17];

    // bf16 transposed weight staging
    convT_bf16_kernel<<<(D_MODEL * D_MODEL + 255) / 256, 256, 0, stream>>>(Wq, wqb, D_MODEL, D_MODEL);
    convT_bf16_kernel<<<(D_MODEL * D_MODEL + 255) / 256, 256, 0, stream>>>(Wk, wkb, D_MODEL, D_MODEL);
    convT_bf16_kernel<<<(D_MODEL * D_MODEL + 255) / 256, 256, 0, stream>>>(Wv, wvb, D_MODEL, D_MODEL);
    convT_bf16_kernel<<<(D_MODEL * D_MODEL + 255) / 256, 256, 0, stream>>>(Wo, wob, D_MODEL, D_MODEL);
    convT_bf16_kernel<<<(D_MODEL * FF_DIM + 255) / 256, 256, 0, stream>>>(W1, w1b, D_MODEL, FF_DIM);
    convT_bf16_kernel<<<(FF_DIM * D_MODEL + 255) / 256, 256, 0, stream>>>(W2, w2b, FF_DIM, D_MODEL);

    // x = LN1(h) -> bf16
    ln_kernel<<<g_rows, 128, 0, stream>>>(hin, nullptr, l1s, l1b, nullptr, xln);
    // Q (pre-scaled), K, V
    gemm_kernel<0><<<g_gemm128, 256, 0, stream>>>(xln, wqb, bq, nullptr, nullptr, Qb,
                                                  D_MODEL, D_MODEL, qscale);
    gemm_kernel<0><<<g_gemm128, 256, 0, stream>>>(xln, wkb, bk, nullptr, nullptr, Kb,
                                                  D_MODEL, D_MODEL, 1.0f);
    gemm_kernel<0><<<g_gemm128, 256, 0, stream>>>(xln, wvb, bv, nullptr, nullptr, Vb,
                                                  D_MODEL, D_MODEL, 1.0f);
    // flash attention with additive key bias
    attn_kernel<<<g_attn, 128, 0, stream>>>(Qb, Kb, Vb, ab, ao);
    // t1 = hin + ao @ Wo + bo
    gemm_kernel<2><<<g_gemm128, 256, 0, stream>>>(ao, wob, bo, hin, t1, nullptr,
                                                  D_MODEL, D_MODEL, 1.0f);
    // x2 = LN2(t1) -> bf16
    ln_kernel<<<g_rows, 128, 0, stream>>>(t1, nullptr, l2s, l2b, nullptr, x2b);
    // u = relu(x2 @ W1 + b1) -> bf16
    gemm_kernel<1><<<g_gemmff, 256, 0, stream>>>(x2b, w1b, b1, nullptr, nullptr, ub,
                                                 D_MODEL, FF_DIM, 1.0f);
    // t2 = t1 + u @ W2 + b2
    gemm_kernel<2><<<g_gemm128, 256, 0, stream>>>(ub, w2b, b2, t1, t2, nullptr,
                                                  FF_DIM, D_MODEL, 1.0f);
    // h_next = LN(hin + t2)   (reference: h + _layer(h))
    ln_kernel<<<g_rows, 128, 0, stream>>>(hin, t2, nsc, nbi, houts[l], nullptr);
    hin = houts[l];
  }
}
